// Head_33285996544367
// MI455X (gfx1250) — matrix operand order repaired and checked by reading, not compiled, not run on hardware
//
#include <hip/hip_runtime.h>

// Single-head causal attention, B=16, T=4096, C=64 (MI455X / gfx1250).
// Kernel 0: pack Wq/Wk/Wv into per-lane WMMA B-operand layout (bf16) once.
// Kernel 1: QKV projection via v_wmma_f32_16x16x32_bf16 -> bf16 Q (pre-scaled),
//           K row-major, V transposed [B][64][T] in d_ws.
// Kernel 2: flash attention; K/V blocks staged into LDS once per workgroup
//           (global_load_async_to_lds_b128 + ASYNCcnt double buffering),
//           S^T = K*Q^T so softmax needs one shfl_xor(16), and P^T reaches the
//           WMMA B layout with a register-only half-wave exchange.

#define BB 16
#define TT 4096
#define CC 64

typedef __attribute__((ext_vector_type(16))) __bf16        v16bf;
typedef __attribute__((ext_vector_type(8)))  __bf16        v8bf;
typedef __attribute__((ext_vector_type(8)))  float         v8f;
typedef __attribute__((ext_vector_type(8)))  unsigned int  v8u;
typedef int vsi4 __attribute__((vector_size(16)));  // matches builtin param type

#if defined(__gfx1250__) && __has_builtin(__builtin_amdgcn_global_load_async_to_lds_b128)
#define HAVE_ASYNC 1
#else
#define HAVE_ASYNC 0
#endif

#if defined(__gfx1250__) && __has_builtin(__builtin_amdgcn_cvt_pk_bf16_f32)
#define HAVE_CVT_PK_BF16 1
#else
#define HAVE_CVT_PK_BF16 0
#endif

// pack two f32 -> packed bf16 dword (a in [15:0], b in [31:16])
static __device__ __forceinline__ unsigned pk2(float a, float b) {
#if HAVE_CVT_PK_BF16
  auto r = __builtin_amdgcn_cvt_pk_bf16_f32(a, b);  // one v_cvt_pk_bf16_f32
  return __builtin_bit_cast(unsigned, r);
#else
  // Let the backend pick the native f32->bf16 conversion for gfx1250.
  unsigned short ua = __builtin_bit_cast(unsigned short, (__bf16)a);
  unsigned short ub = __builtin_bit_cast(unsigned short, (__bf16)b);
  return (unsigned)ua | ((unsigned)ub << 16);
#endif
}
static __device__ __forceinline__ unsigned short f2bf1(float a) {
  return __builtin_bit_cast(unsigned short, (__bf16)a);
}
static __device__ __forceinline__ v8f wmma_bf16(v16bf a, v16bf b, v8f c) {
  return __builtin_amdgcn_wmma_f32_16x16x32_bf16(false, a, false, b, (short)0, c,
                                                 false, false);
}
static __device__ __forceinline__ v16bf ld_a16(const __bf16* p0, const __bf16* p1) {
  v8bf a = *(const v8bf*)p0;
  v8bf b = *(const v8bf*)p1;
  return __builtin_shufflevector(a, b, 0, 1, 2, 3, 4, 5, 6, 7,
                                 8, 9, 10, 11, 12, 13, 14, 15);
}
static __device__ __forceinline__ void cp16_async(const unsigned short* g,
                                                  unsigned short* l) {
#if HAVE_ASYNC
  vsi4* gp = (vsi4*)g;  // strip const, type as int4-vector
  vsi4* lp = (vsi4*)l;
  __builtin_amdgcn_global_load_async_to_lds_b128(
      (__attribute__((address_space(1))) vsi4*)gp,
      (__attribute__((address_space(3))) vsi4*)lp, 0, 0);
#else
  *(uint4*)l = *(const uint4*)g;
#endif
}
static __device__ __forceinline__ void wait_async0() {
#if defined(__gfx1250__)
#if __has_builtin(__builtin_amdgcn_s_wait_asynccnt)
  __builtin_amdgcn_s_wait_asynccnt(0);
#else
  asm volatile("s_wait_asynccnt 0" ::: "memory");
#endif
#endif
}

// ---------------- Kernel 0: pack weights into WMMA B-operand layout ----------
// Wpk dword index = combo*256 + lane*8 + j, combo = ((m*4+t)*2+s), m in {q,k,v}.
__global__ __launch_bounds__(256) void pack_w_kernel(
    const float* __restrict__ Wq, const float* __restrict__ Wk,
    const float* __restrict__ Wv, unsigned int* __restrict__ Wpk) {
  const float* Ws[3] = {Wq, Wk, Wv};
  for (int idx = threadIdx.x; idx < 24 * 32 * 8; idx += 256) {
    const int j     = idx & 7;
    const int lane  = (idx >> 3) & 31;
    const int combo = idx >> 8;
    const int s  = combo & 1;
    const int t  = (combo >> 1) & 3;
    const int m  = combo >> 3;
    const int n  = lane & 15;
    const int hi = lane >> 4;
    const int cbase = 32 * s + (hi ? 16 : 0);  // B layout: lo lanes K=0..15
    const int jcol  = 16 * t + n;
    const float* W = Ws[m];
    Wpk[idx] = pk2(W[(cbase + 2 * j) * CC + jcol],
                   W[(cbase + 2 * j + 1) * CC + jcol]);
  }
}

// ---------------- Kernel 1: QKV projection ----------------
__global__ __launch_bounds__(256) void qkv_proj_kernel(
    const float* __restrict__ x, const unsigned int* __restrict__ Wpk,
    unsigned short* __restrict__ Qs, unsigned short* __restrict__ Ksb,
    unsigned short* __restrict__ Vts) {
  const int lane = threadIdx.x & 31;
  const int wv   = threadIdx.x >> 5;
  const int n    = lane & 15;
  const bool hi  = lane >= 16;
  const int r0   = blockIdx.x * 128 + wv * 16;

  // X rows r0..r0+15 in WMMA A layout (bf16), K dim = 64 -> 2 k-steps.
  const float* xr = x + (size_t)(r0 + n) * CC;
  v16bf Xa[2];
#pragma unroll
  for (int s = 0; s < 2; ++s) {
    const int c0 = 32 * s + (hi ? 16 : 0);
    const int c1 = c0 + 8;
    float4 f0 = *(const float4*)(xr + c0);
    float4 f1 = *(const float4*)(xr + c0 + 4);
    float4 f2 = *(const float4*)(xr + c1);
    float4 f3 = *(const float4*)(xr + c1 + 4);
    v8u u;
    u[0] = pk2(f0.x, f0.y); u[1] = pk2(f0.z, f0.w);
    u[2] = pk2(f1.x, f1.y); u[3] = pk2(f1.z, f1.w);
    u[4] = pk2(f2.x, f2.y); u[5] = pk2(f2.z, f2.w);
    u[6] = pk2(f3.x, f3.y); u[7] = pk2(f3.z, f3.w);
    Xa[s] = __builtin_bit_cast(v16bf, u);
  }

  const v16bf* Wt = (const v16bf*)Wpk;
#pragma unroll
  for (int m = 0; m < 3; ++m) {
#pragma unroll
    for (int t = 0; t < 4; ++t) {
      const int cb = (m * 4 + t) * 2;
      v16bf Wb0 = Wt[cb * 32 + lane];
      v16bf Wb1 = Wt[(cb + 1) * 32 + lane];
      v8f acc = {};
      acc = wmma_bf16(Xa[0], Wb0, acc);
      acc = wmma_bf16(Xa[1], Wb1, acc);
      const int jcol = 16 * t + n;
      const int rowb = r0 + (hi ? 8 : 0);
#pragma unroll
      for (int r = 0; r < 8; ++r) {
        const int row = rowb + r;
        if (m == 0) {
          Qs[(size_t)row * CC + jcol] = f2bf1(acc[r] * 0.125f);  // fold C^-0.5
        } else if (m == 1) {
          Ksb[(size_t)row * CC + jcol] = f2bf1(acc[r]);
        } else {
          const int bb   = row >> 12;  // / T
          const int trow = row & (TT - 1);
          Vts[(size_t)bb * CC * TT + (size_t)jcol * TT + trow] = f2bf1(acc[r]);
        }
      }
    }
  }
}

// ---------------- Kernel 2: flash attention ----------------
// grid: B*(T/128) blocks of 256 threads (8 waves x 16 queries). K/V 64-key
// blocks staged through LDS (72-element padded rows -> conflict-free b128 reads).
__global__ __launch_bounds__(256) void flash_attn_kernel(
    const unsigned short* __restrict__ Qsu, const unsigned short* __restrict__ Ksu,
    const unsigned short* __restrict__ Vtu, float* __restrict__ out) {
  __shared__ __align__(16) unsigned short KL[2][64][72];
  __shared__ __align__(16) unsigned short VL[2][64][72];

  const int tid  = threadIdx.x;
  const int lane = tid & 31;
  const int wv   = tid >> 5;
  const int n    = lane & 15;
  const bool hi  = lane >= 16;
  const int b    = blockIdx.x >> 5;  // 32 query blocks per batch
  const int qblk = blockIdx.x & 31;
  const int qw   = qblk * 128 + wv * 16;
  const int qidx = qw + n;
  const int h8   = hi ? 8 : 0;

  const unsigned short* Kbase = Ksu + (size_t)b * TT * CC;
  const unsigned short* Vbase = Vtu + (size_t)b * CC * TT;

  // Q^T in B layout: lane = query column, dims packed along VGPRs.
  const __bf16* qrow = (const __bf16*)Qsu + (size_t)(b * TT + qidx) * CC;
  v16bf Qb[2];
#pragma unroll
  for (int s = 0; s < 2; ++s)
    Qb[s] = *(const v16bf*)(qrow + 32 * s + (hi ? 16 : 0));

  float mrun = -3.0e38f;
  float lrun = 0.0f;
  v8f o[4] = {};

  // cooperative staging of one 64-key block (K: 64x64, V^T: 64x64, bf16)
  auto stage_issue = [&](int buf, int k0) {
#pragma unroll
    for (int i = 0; i < 2; ++i) {
      const int cc = tid + 256 * i;  // 0..511 16B-chunks
      const int r  = cc >> 3;
      const int c  = (cc & 7) * 8;
      cp16_async(Kbase + (size_t)(k0 + r) * CC + c, &KL[buf][r][c]);
      cp16_async(Vbase + (size_t)r * TT + k0 + c, &VL[buf][r][c]);
    }
  };
  auto stage_sync = [&](int buf, int k0) {
#pragma unroll
    for (int i = 0; i < 2; ++i) {
      const int cc = tid + 256 * i;
      const int r  = cc >> 3;
      const int c  = (cc & 7) * 8;
      uint4 kk = *(const uint4*)(Kbase + (size_t)(k0 + r) * CC + c);
      uint4 vv = *(const uint4*)(Vbase + (size_t)r * TT + k0 + c);
      *(uint4*)&KL[buf][r][c] = kk;
      *(uint4*)&VL[buf][r][c] = vv;
    }
  };

  const int KB = 2 * qblk + 1;  // uniform causal bound for the whole block

#if HAVE_ASYNC
  stage_issue(0, 0);
#endif
  for (int kb = 0; kb <= KB; ++kb) {
    const int cur = kb & 1;
    const int k0  = kb * 64;
#if HAVE_ASYNC
    wait_async0();    // my share of buffer `cur` landed
    __syncthreads();  // everyone's share landed; prev compute finished
    if (kb < KB) stage_issue(cur ^ 1, k0 + 64);  // overlap next DMA with compute
#else
    __syncthreads();
    stage_sync(cur, k0);
    __syncthreads();
#endif

    if (k0 <= qw + 15) {  // skip fully-masked blocks (still hit barriers)
      // ---- S^T = K * Q^T : 4 tiles of 16 keys x 16 queries ----
      v8f st[4];
#pragma unroll
      for (int t = 0; t < 4; ++t) {
        const __bf16* krow = (const __bf16*)&KL[cur][16 * t + n][0];
        v16bf a0 = ld_a16(krow + 2 * h8, krow + 8 + 2 * h8);
        v16bf a1 = ld_a16(krow + 32 + 2 * h8, krow + 40 + 2 * h8);
        v8f c = {};
        c = wmma_bf16(a0, Qb[0], c);
        c = wmma_bf16(a1, Qb[1], c);
        st[t] = c;
      }

      // ---- causal mask (diagonal block only) ----
      if (k0 + 63 > qw) {
        const int kbase0 = k0 + h8;
#pragma unroll
        for (int t = 0; t < 4; ++t)
#pragma unroll
          for (int r = 0; r < 8; ++r)
            if (kbase0 + 16 * t + r > qidx) st[t][r] = -1.0e30f;
      }

      // ---- online softmax (per query == per lane column) ----
      float mloc = -3.0e38f;
#pragma unroll
      for (int t = 0; t < 4; ++t)
#pragma unroll
        for (int r = 0; r < 8; ++r) mloc = fmaxf(mloc, st[t][r]);
      mloc = fmaxf(mloc, __shfl_xor(mloc, 16, 32));
      const float mnew  = fmaxf(mrun, mloc);
      const float alpha = __expf(mrun - mnew);
      float psum = 0.0f;
#pragma unroll
      for (int t = 0; t < 4; ++t)
#pragma unroll
        for (int r = 0; r < 8; ++r) {
          const float p = __expf(st[t][r] - mnew);
          st[t][r] = p;
          psum += p;
        }
      psum += __shfl_xor(psum, 16, 32);
      lrun = lrun * alpha + psum;
      mrun = mnew;
#pragma unroll
      for (int t = 0; t < 4; ++t)
#pragma unroll
        for (int r = 0; r < 8; ++r) o[t][r] *= alpha;

      // ---- P^T (C/D layout) -> B layout via half-wave exchange ----
      v16bf Pb[2];
#pragma unroll
      for (int g = 0; g < 2; ++g) {
        v8u u;
#pragma unroll
        for (int j = 0; j < 4; ++j) {
          const unsigned xp = pk2(st[2 * g][2 * j], st[2 * g][2 * j + 1]);
          const unsigned zp = pk2(st[2 * g + 1][2 * j], st[2 * g + 1][2 * j + 1]);
          const unsigned yp = __shfl_xor(xp, 16, 32);
          const unsigned wp = __shfl_xor(zp, 16, 32);
          u[j]     = hi ? wp : xp;  // lo: keys 2j,2j+1   | hi: keys 16+2j..
          u[4 + j] = hi ? zp : yp;  // lo: keys 8+2j..    | hi: keys 24+2j..
        }
        Pb[g] = __builtin_bit_cast(v16bf, u);
      }

      // ---- O^T += V^T * P^T : 4 head-dim tiles ----
#pragma unroll
      for (int td = 0; td < 4; ++td) {
        const __bf16* vrow = (const __bf16*)&VL[cur][16 * td + n][0];
        v16bf a0 = ld_a16(vrow + 2 * h8, vrow + 8 + 2 * h8);
        v16bf a1 = ld_a16(vrow + 32 + 2 * h8, vrow + 40 + 2 * h8);
        o[td] = wmma_bf16(a0, Pb[0], o[td]);
        o[td] = wmma_bf16(a1, Pb[1], o[td]);
      }
    }
  }

  // ---- epilogue: out[b][q][d] = O^T[d][q] / l ----
  const float rl = 1.0f / lrun;
  float* op = out + (size_t)(b * TT + qidx) * CC;
#pragma unroll
  for (int td = 0; td < 4; ++td) {
    float4 lo4 = make_float4(o[td][0] * rl, o[td][1] * rl, o[td][2] * rl, o[td][3] * rl);
    float4 hi4 = make_float4(o[td][4] * rl, o[td][5] * rl, o[td][6] * rl, o[td][7] * rl);
    *(float4*)(op + 16 * td + h8)     = lo4;
    *(float4*)(op + 16 * td + h8 + 4) = hi4;
  }
}

extern "C" void kernel_launch(void* const* d_in, const int* in_sizes, int n_in,
                              void* d_out, int out_size, void* d_ws, size_t ws_size,
                              hipStream_t stream) {
  (void)in_sizes; (void)n_in; (void)out_size; (void)ws_size;
  const float* x  = (const float*)d_in[0];
  const float* Wq = (const float*)d_in[1];
  const float* Wk = (const float*)d_in[2];
  const float* Wv = (const float*)d_in[3];
  float* out = (float*)d_out;

  const size_t N = (size_t)BB * TT * CC;           // 4,194,304 elements
  unsigned short* Qs  = (unsigned short*)d_ws;     // bf16 Q   (8 MB)
  unsigned short* Ks  = Qs + N;                    // bf16 K   (8 MB)
  unsigned short* Vts = Ks + N;                    // bf16 V^T (8 MB)
  unsigned int*   Wpk = (unsigned int*)(Vts + N);  // packed W operands (24 KB)

  pack_w_kernel<<<dim3(1), dim3(256), 0, stream>>>(Wq, Wk, Wv, Wpk);
  qkv_proj_kernel<<<dim3((BB * TT) / 128), dim3(256), 0, stream>>>(
      x, Wpk, Qs, Ks, Vts);
  flash_attn_kernel<<<dim3(BB * (TT / 128)), dim3(256), 0, stream>>>(
      Qs, Ks, Vts, out);
}
